// CondMV2Transformer_75505525063710
// MI455X (gfx1250) — compile-verified, matrix operand order checked by reading, not run on hardware
//
#include <hip/hip_runtime.h>
#include <cstdint>
#include <cstddef>

// ---------------- model constants ----------------
#define D_    256
#define H_    8
#define HD_   32
#define NSEQ  2048
#define BB    2
#define FH_   682
#define VOCAB 32000
#define NOFF  44
#define MTOK  (BB * NSEQ)   // 4096 tokens

typedef __bf16 bf16_t;
typedef bf16_t v16bf __attribute__((ext_vector_type(16)));
typedef float  v8f   __attribute__((ext_vector_type(8)));
typedef unsigned int u32x4 __attribute__((ext_vector_type(4)));
typedef int i32x8 __attribute__((ext_vector_type(8)));
typedef int i32x4 __attribute__((ext_vector_type(4)));

__constant__ int c_offs[NOFF] = {
    0,1,2,3,4,5,6,7,8,9,10,11,12,13,14,15,16,17,18,19,20,21,22,23,24,25,26,
    27,28,29,30,31,32,48,64,96,128,192,256,384,512,768,1024,1536};

// ---------------------------------------------------------------------------
// Embedding gather: x[token_row, :] = emb[token]
// ---------------------------------------------------------------------------
__global__ void embed_kernel(const int* __restrict__ tokens,
                             const float* __restrict__ emb,
                             float* __restrict__ x) {
  int row = blockIdx.x;           // 0..MTOK-1
  int d   = threadIdx.x;          // 0..255
  x[(size_t)row * D_ + d] = emb[(size_t)tokens[row] * D_ + d];
}

// ---------------------------------------------------------------------------
// RMSNorm: one 256-thread block per token (D_ == 256)
// ---------------------------------------------------------------------------
__global__ void rmsnorm_kernel(const float* __restrict__ x,
                               const float* __restrict__ s,
                               float* __restrict__ y) {
  __shared__ float red[D_];
  int row = blockIdx.x;
  int d   = threadIdx.x;
  float v = x[(size_t)row * D_ + d];
  red[d] = v * v;
  __syncthreads();
  for (int off = D_ / 2; off > 0; off >>= 1) {
    if (d < off) red[d] += red[d + off];
    __syncthreads();
  }
  float scale = rsqrtf(red[0] * (1.0f / D_) + 1e-6f);
  y[(size_t)row * D_ + d] = s[d] * v * scale;
}

// ---------------------------------------------------------------------------
// Generic WMMA bf16 GEMM: C[M,N] = epi(A[M,K] @ W[K,N])
//   A, W are fp32 in HBM, converted to bf16 while staging into LDS,
//   fp32 accumulate in v_wmma_f32_16x16x32_bf16.
// Block: 256 threads (8 wave32's), tile 128(M) x 64(N), K-step 32.
// Each wave owns a 32x32 tile = 2x2 WMMA 16x16 tiles.
// EPI: 0=none 1=sigmoid(v+bias) 2=silu(v) 3=v+res 4=v+bias
// ---------------------------------------------------------------------------
template <int EPI>
__global__ __launch_bounds__(256) void gemm_kernel(
    const float* __restrict__ A, const float* __restrict__ W,
    const float* __restrict__ bias, const float* __restrict__ res,
    float* __restrict__ C, int M, int N, int K) {
  __shared__ bf16_t sA[128][34];  // +pad to break bank conflicts
  __shared__ bf16_t sB[32][72];

  const int tid  = threadIdx.x;
  const int lane = tid & 31;
  const int wave = tid >> 5;
  const int ml   = lane & 15;
  const int hf   = lane >> 4;              // lane half (0/1)
  const int wm   = (wave & 3) * 32;        // wave M offset in block tile
  const int wn   = (wave >> 2) * 32;       // wave N offset in block tile
  const int bm   = blockIdx.y * 128;
  const int bn   = blockIdx.x * 64;

  v8f vzero = {};
  v8f acc[2][2];
#pragma unroll
  for (int s = 0; s < 2; ++s)
#pragma unroll
    for (int t = 0; t < 2; ++t) acc[s][t] = vzero;

  for (int k0 = 0; k0 < K; k0 += 32) {
    // ---- stage A tile (128x32 fp32 -> bf16 LDS), 16 elems/thread ----
#pragma unroll
    for (int j = 0; j < 16; ++j) {
      int lin = j * 256 + tid;
      int r = lin >> 5, c = lin & 31;
      int gr = bm + r, gc = k0 + c;
      float v = 0.0f;
      if (gr < M && gc < K) v = A[(size_t)gr * K + gc];
      sA[r][c] = (bf16_t)v;
    }
    // ---- stage B tile (32x64 fp32 -> bf16 LDS), 8 elems/thread ----
#pragma unroll
    for (int j = 0; j < 8; ++j) {
      int lin = j * 256 + tid;
      int r = lin >> 6, c = lin & 63;
      int gr = k0 + r, gc = bn + c;
      float v = 0.0f;
      if (gr < K && gc < N) v = W[(size_t)gr * N + gc];
      sB[r][c] = (bf16_t)v;
    }
    if (k0 + 32 < K)  // emits global_prefetch_b8 for next weight K-slice
      __builtin_prefetch(&W[(size_t)(k0 + 32) * N + bn], 0, 1);
    __syncthreads();

    // ---- build fragments per the CDNA5 16-bit A/B VGPR layout ----
    v16bf af[2], bfv[2];
#pragma unroll
    for (int s = 0; s < 2; ++s) {
      int mrow = wm + s * 16 + ml;
#pragma unroll
      for (int i = 0; i < 8; ++i) {
        int kk = ((i < 4) ? 2 * i : 16 + 2 * (i - 4)) + hf * 8;
        af[s][2 * i]     = sA[mrow][kk];
        af[s][2 * i + 1] = sA[mrow][kk + 1];
      }
    }
#pragma unroll
    for (int t = 0; t < 2; ++t) {
      int ncol = wn + t * 16 + ml;
#pragma unroll
      for (int i = 0; i < 8; ++i) {
        int kk = ((i < 4) ? 2 * i : 16 + 2 * (i - 4)) + hf * 8;
        bfv[t][2 * i]     = sB[kk][ncol];
        bfv[t][2 * i + 1] = sB[kk + 1][ncol];
      }
    }
#pragma unroll
    for (int s = 0; s < 2; ++s)
#pragma unroll
      for (int t = 0; t < 2; ++t)
        acc[s][t] = __builtin_amdgcn_wmma_f32_16x16x32_bf16(
            false, af[s], false, bfv[t], (short)0, acc[s][t], false, false);
    __syncthreads();
  }

  // ---- epilogue (C layout: VGPR r -> row r + 8*hf, col = ml) ----
#pragma unroll
  for (int s = 0; s < 2; ++s)
#pragma unroll
    for (int t = 0; t < 2; ++t)
#pragma unroll
      for (int r = 0; r < 8; ++r) {
        int row = bm + wm + s * 16 + r + hf * 8;
        int col = bn + wn + t * 16 + ml;
        if (row < M && col < N) {
          float v = acc[s][t][r];
          if (EPI == 1)      v = 1.0f / (1.0f + __expf(-(v + bias[col])));
          else if (EPI == 2) v = v * (1.0f / (1.0f + __expf(-v)));
          else if (EPI == 3) v = v + res[(size_t)row * N + col];
          else if (EPI == 4) v = v + bias[col];
          C[(size_t)row * N + col] = v;
        }
      }
}

// ---------------------------------------------------------------------------
// Dynamic-sparse (shifted offsets) attention. O=44 offsets, HD=32: too skinny
// for WMMA, so one VALU thread per (b,h,t).
// qkv layout: [tok, 768] with q=c, k=256+c, v=512+c, c = h*32+d.
// ---------------------------------------------------------------------------
__global__ void dsqg_kernel(const float* __restrict__ qkv,
                            const float* __restrict__ pb,  // [O,H]
                            float* __restrict__ attn) {
  int gid = blockIdx.x * blockDim.x + threadIdx.x;
  if (gid >= BB * H_ * NSEQ) return;
  int t = gid & (NSEQ - 1);
  int h = (gid >> 11) & (H_ - 1);
  int b = gid >> 14;

  const float scale = 0.17677669529663687f;  // 1/sqrt(32)
  size_t rowbase = ((size_t)(b * NSEQ + t)) * (3 * D_) + h * HD_;

  float q[HD_];
#pragma unroll
  for (int d = 0; d < HD_; d += 4) {
    float4 v4 = *(const float4*)(qkv + rowbase + d);
    q[d] = v4.x; q[d + 1] = v4.y; q[d + 2] = v4.z; q[d + 3] = v4.w;
  }

  float sc[NOFF];
  float mx = -1e30f;
  for (int o = 0; o < NOFF; ++o) {
    int idx = t - c_offs[o];
    if (idx < 0) { sc[o] = -1e30f; continue; }
    const float* kp = qkv + ((size_t)(b * NSEQ + idx)) * (3 * D_) + D_ + h * HD_;
    float dot = 0.0f;
#pragma unroll
    for (int d = 0; d < HD_; ++d) dot += q[d] * kp[d];
    float s = dot * scale + pb[o * H_ + h];
    sc[o] = s;
    mx = fmaxf(mx, s);
  }
  float sum = 0.0f;
  for (int o = 0; o < NOFF; ++o) {
    float e = (sc[o] <= -1e29f) ? 0.0f : __expf(sc[o] - mx);
    sc[o] = e;
    sum += e;
  }
  float inv = 1.0f / sum;
  float out[HD_];
#pragma unroll
  for (int d = 0; d < HD_; ++d) out[d] = 0.0f;
  for (int o = 0; o < NOFF; ++o) {
    if (sc[o] == 0.0f) continue;
    int idx = t - c_offs[o];
    const float* vp = qkv + ((size_t)(b * NSEQ + idx)) * (3 * D_) + 2 * D_ + h * HD_;
    float w = sc[o] * inv;
#pragma unroll
    for (int d = 0; d < HD_; ++d) out[d] += w * vp[d];
  }
  size_t ob = ((size_t)(b * NSEQ + t)) * D_ + h * HD_;
#pragma unroll
  for (int d = 0; d < HD_; ++d) attn[ob + d] = out[d];
}

// ---------------------------------------------------------------------------
// RoPE (layer 5 only), in place on q,k halves of qkv. One thread per pair.
// ---------------------------------------------------------------------------
__global__ void rope_kernel(float* __restrict__ qkv) {
  int gid = blockIdx.x * blockDim.x + threadIdx.x;  // BB*NSEQ*H*16
  if (gid >= BB * NSEQ * H_ * 16) return;
  int j = gid & 15;
  int h = (gid >> 4) & (H_ - 1);
  int t = (gid >> 7) & (NSEQ - 1);
  int b = gid >> 18;
  float invf = __powf(10000.0f, -(float)(2 * j) / (float)HD_);
  float ang = (float)t * invf;
  float c = __cosf(ang), sn = __sinf(ang);
  size_t base = ((size_t)(b * NSEQ + t)) * (3 * D_) + h * HD_;
  float q1 = qkv[base + j], q2 = qkv[base + j + 16];
  qkv[base + j]      = q1 * c - q2 * sn;
  qkv[base + j + 16] = q2 * c + q1 * sn;
  float k1 = qkv[base + D_ + j], k2 = qkv[base + D_ + j + 16];
  qkv[base + D_ + j]      = k1 * c - k2 * sn;
  qkv[base + D_ + j + 16] = k2 * c + k1 * sn;
}

// ---------------------------------------------------------------------------
// Flash attention for the full causal layer, WMMA-based.
// One wave (32 threads) per (b, h, 16-query tile). HD=32 == one bf16 K-step,
// so S(16x16) = one v_wmma per key tile; P x V through LDS layout conversion.
// ---------------------------------------------------------------------------
__global__ __launch_bounds__(32) void flash_kernel(const float* __restrict__ qkv,
                                                   float* __restrict__ attn) {
  __shared__ bf16_t sP[16][24];
  __shared__ bf16_t sV[16][40];

  int blk = blockIdx.x;               // b*H*(N/16): qt fastest
  int qt = blk & (NSEQ / 16 - 1);     // 128 tiles
  int h  = (blk >> 7) & (H_ - 1);
  int b  = blk >> 10;
  int lane = threadIdx.x;
  int ml = lane & 15;
  int hf = lane >> 4;

  const float scale = 0.17677669529663687f;

  // Q fragment (A layout): lane row = ml
  v16bf aQ;
  {
    int qrow = qt * 16 + ml;
    size_t qbase = ((size_t)(b * NSEQ + qrow)) * (3 * D_) + h * HD_;
#pragma unroll
    for (int i = 0; i < 8; ++i) {
      int kk = ((i < 4) ? 2 * i : 16 + 2 * (i - 4)) + hf * 8;
      aQ[2 * i]     = (bf16_t)qkv[qbase + kk];
      aQ[2 * i + 1] = (bf16_t)qkv[qbase + kk + 1];
    }
  }

  float Mrow[8], Lrow[8];
  v8f acc0 = {}, acc1 = {};
#pragma unroll
  for (int r = 0; r < 8; ++r) { Mrow[r] = -1e30f; Lrow[r] = 0.0f; }

  for (int kt = 0; kt <= qt; ++kt) {
    // K^T fragment (B layout, 32(d) x 16(key)): lane col = key = kt*16+ml
    v16bf bK;
    {
      int key = kt * 16 + ml;
      size_t kbase = ((size_t)(b * NSEQ + key)) * (3 * D_) + D_ + h * HD_;
#pragma unroll
      for (int i = 0; i < 8; ++i) {
        int kk = ((i < 4) ? 2 * i : 16 + 2 * (i - 4)) + hf * 8;
        bK[2 * i]     = (bf16_t)qkv[kbase + kk];
        bK[2 * i + 1] = (bf16_t)qkv[kbase + kk + 1];
      }
    }
    v8f S = {};
    S = __builtin_amdgcn_wmma_f32_16x16x32_bf16(false, aQ, false, bK, (short)0,
                                                S, false, false);
    // online softmax (C layout: elem r -> row r+8*hf, col ml)
#pragma unroll
    for (int r = 0; r < 8; ++r) {
      float s = S[r] * scale;
      if (kt == qt && ml > r + hf * 8) s = -1e30f;  // causal mask in diag tile
      float rm = s;
#pragma unroll
      for (int m = 1; m < 16; m <<= 1) rm = fmaxf(rm, __shfl_xor(rm, m));
      float newM = fmaxf(Mrow[r], rm);
      float f = __expf(Mrow[r] - newM);
      float p = __expf(s - newM);
      float rs = p;
#pragma unroll
      for (int m = 1; m < 16; m <<= 1) rs += __shfl_xor(rs, m);
      Lrow[r] = Lrow[r] * f + rs;
      acc0[r] *= f;
      acc1[r] *= f;
      Mrow[r] = newM;
      sP[r + hf * 8][ml] = (bf16_t)p;  // C layout -> LDS tile
    }
    // stage V tile (16 keys x 32 dims) to LDS
#pragma unroll
    for (int j = 0; j < 16; ++j) {
      int lin = j * 32 + lane;
      int krow = lin >> 5, d = lin & 31;
      sV[krow][d] = (bf16_t)qkv[((size_t)(b * NSEQ + kt * 16 + krow)) * (3 * D_) +
                                2 * D_ + h * HD_ + d];
    }
    __syncthreads();
    // P (A-frag, K padded 16->32 with zeros) and V (B-frags for d 0..15,16..31)
    v16bf aP, bV0, bV1;
    bf16_t z = (bf16_t)0.0f;
#pragma unroll
    for (int i = 0; i < 8; ++i) {
      int kk = ((i < 4) ? 2 * i : 16 + 2 * (i - 4)) + hf * 8;
      int kk1 = kk + 1;
      aP[2 * i]      = (kk < 16) ? sP[ml][kk] : z;
      aP[2 * i + 1]  = (kk1 < 16) ? sP[ml][kk1] : z;
      bV0[2 * i]     = (kk < 16) ? sV[kk][ml] : z;
      bV0[2 * i + 1] = (kk1 < 16) ? sV[kk1][ml] : z;
      bV1[2 * i]     = (kk < 16) ? sV[kk][16 + ml] : z;
      bV1[2 * i + 1] = (kk1 < 16) ? sV[kk1][16 + ml] : z;
    }
    acc0 = __builtin_amdgcn_wmma_f32_16x16x32_bf16(false, aP, false, bV0,
                                                   (short)0, acc0, false, false);
    acc1 = __builtin_amdgcn_wmma_f32_16x16x32_bf16(false, aP, false, bV1,
                                                   (short)0, acc1, false, false);
    __syncthreads();
  }
#pragma unroll
  for (int r = 0; r < 8; ++r) {
    int row = qt * 16 + r + hf * 8;
    float invL = 1.0f / Lrow[r];
    size_t ob = ((size_t)(b * NSEQ + row)) * D_ + h * HD_;
    attn[ob + ml]      = acc0[r] * invL;
    attn[ob + 16 + ml] = acc1[r] * invL;
  }
}

// ---------------------------------------------------------------------------
// Pointwise helpers
// ---------------------------------------------------------------------------
__global__ void mul_kernel(float* __restrict__ y, const float* __restrict__ g,
                           int n) {
  int i = blockIdx.x * blockDim.x + threadIdx.x;
  if (i < n) y[i] *= g[i];
}
__global__ void gated_add_kernel(float* __restrict__ x,
                                 const float* __restrict__ g,
                                 const float* __restrict__ t, int n) {
  int i = blockIdx.x * blockDim.x + threadIdx.x;
  if (i < n) x[i] += g[i] * t[i];
}
// causal running mean over sequence: one thread per (b, d)
__global__ void cumsum_pool_kernel(const float* __restrict__ xi,
                                   float* __restrict__ pool) {
  int gid = blockIdx.x * blockDim.x + threadIdx.x;
  if (gid >= BB * D_) return;
  int b = gid >> 8;
  int d = gid & (D_ - 1);
  float s = 0.0f;
  for (int n = 0; n < NSEQ; ++n) {
    size_t idx = ((size_t)(b * NSEQ + n)) * D_ + d;
    s += xi[idx];
    pool[idx] = s / (float)(n + 1);
  }
}

// ---------------------------------------------------------------------------
// TDM warm kernel: demonstrates the CDNA5 Tensor Data Mover path.
// Issues tensor_load_to_lds of a 64x16 fp32 tile of the LM head into LDS,
// then s_wait_tensorcnt 0. (Descriptor per CDNA5 ISA.)
// This toolchain exposes the 6-arg builtin: (g0,g1,g2,g3,g4,cpol).
// ---------------------------------------------------------------------------
__global__ __launch_bounds__(32) void tdm_warm_kernel(const float* W) {
#if defined(__AMDGCN__) && __has_builtin(__builtin_amdgcn_tensor_load_to_lds)
  __shared__ __align__(128) unsigned char lds_buf[4096];
  if (threadIdx.x == 0) {
    unsigned lds_addr = (unsigned)(uintptr_t)lds_buf;
    unsigned long long ga = (unsigned long long)(uintptr_t)W;
    u32x4 g0;
    g0[0] = 1u;                                    // count=1, user descriptor
    g0[1] = lds_addr;                              // lds_addr [63:32]
    g0[2] = (unsigned)(ga & 0xffffffffu);          // global_addr low
    g0[3] = (unsigned)((ga >> 32) & 0x01ffffffu) | (2u << 30);  // type=2
    i32x8 g1;
    g1[0] = 2 << 16;                               // data_size=4B, mask=0
    g1[1] = (VOCAB & 0xffff) << 16;                // tensor_dim0 lo16
    g1[2] = (VOCAB >> 16) | ((D_ & 0xffff) << 16); // dim0 hi / dim1 lo
    g1[3] = (64 << 16);                            // tile_dim0 = 64
    g1[4] = 16;                                    // tile_dim1 = 16
    g1[5] = VOCAB;                                 // tensor_dim0_stride lo32
    g1[6] = 0;
    g1[7] = 0;
    i32x4 g2 = {0, 0, 0, 0};                       // dims 2/3 unused (2D)
    i32x4 g3 = {0, 0, 0, 0};
    i32x8 g4 = {0, 0, 0, 0, 0, 0, 0, 0};           // extra group (6-arg form)
    __builtin_amdgcn_tensor_load_to_lds(g0, g1, g2, g3, g4, 0);
  }
#if __has_builtin(__builtin_amdgcn_s_wait_tensorcnt)
  __builtin_amdgcn_s_wait_tensorcnt(0);
#endif
  __syncthreads();
#else
  (void)W;
#endif
}

// ---------------------------------------------------------------------------
// Host side
// ---------------------------------------------------------------------------
static inline void launch_gemm(int epi, const float* A, const float* W,
                               const float* bias, const float* res, float* C,
                               int M, int N, int K, hipStream_t s) {
  dim3 grid((unsigned)((N + 63) / 64), (unsigned)((M + 127) / 128));
  dim3 blk(256);
  switch (epi) {
    case 0: gemm_kernel<0><<<grid, blk, 0, s>>>(A, W, bias, res, C, M, N, K); break;
    case 1: gemm_kernel<1><<<grid, blk, 0, s>>>(A, W, bias, res, C, M, N, K); break;
    case 2: gemm_kernel<2><<<grid, blk, 0, s>>>(A, W, bias, res, C, M, N, K); break;
    case 3: gemm_kernel<3><<<grid, blk, 0, s>>>(A, W, bias, res, C, M, N, K); break;
    case 4: gemm_kernel<4><<<grid, blk, 0, s>>>(A, W, bias, res, C, M, N, K); break;
  }
}

struct LayerP {
  const float *ffn_d, *ffn_g, *ffn_u, *gate_b, *gate_w;
  const float *ig_b, *ig_w, *inter_norm, *ip_b, *ip_w;
  const float *norm1, *norm2, *out, *pos_bias, *qkv;
};

extern "C" void kernel_launch(void* const* d_in, const int* in_sizes, int n_in,
                              void* d_out, int out_size, void* d_ws,
                              size_t ws_size, hipStream_t stream) {
  (void)in_sizes; (void)n_in; (void)out_size;
  // ---- input mapping: jax pytree flatten order (dict keys sorted) ----
  // 0: embedding, 1: final_norm, 2..: layers (each sorted keys), then
  // lm_head, tokens.
  auto F = [&](int i) { return (const float*)d_in[i]; };
  const float* embedding  = F(0);
  const float* final_norm = F(1);
  LayerP L[6];
  int p = 2;
  for (int i = 0; i < 6; ++i) {
    L[i].ffn_d = F(p++); L[i].ffn_g = F(p++); L[i].ffn_u = F(p++);
    L[i].gate_b = F(p++); L[i].gate_w = F(p++);
    if (i == 2) {
      L[i].ig_b = F(p++); L[i].ig_w = F(p++); L[i].inter_norm = F(p++);
      L[i].ip_b = F(p++); L[i].ip_w = F(p++);
    } else {
      L[i].ig_b = L[i].ig_w = L[i].inter_norm = L[i].ip_b = L[i].ip_w = nullptr;
    }
    L[i].norm1 = F(p++); L[i].norm2 = F(p++); L[i].out = F(p++);
    L[i].pos_bias = (i != 5) ? F(p++) : nullptr;
    L[i].qkv = F(p++);
  }
  const float* lm_head = F(p++);
  const int* tokens = (const int*)d_in[p++];

  // ---- workspace layout (floats) ----
  const size_t SZ_X = (size_t)MTOK * D_;        // 1,048,576
  const size_t SZ_Q = (size_t)MTOK * 3 * D_;    // 3,145,728
  const size_t SZ_F = (size_t)MTOK * FH_;       // 2,793,472
  const size_t NEED = (5 * SZ_X + SZ_Q + 2 * SZ_F) * sizeof(float);
  if (ws_size < NEED || d_out == nullptr) return;
  float* wsf   = (float*)d_ws;
  float* x     = wsf;
  float* xn    = x + SZ_X;
  float* attn  = xn + SZ_X;
  float* gateb = attn + SZ_X;
  float* tmpb  = gateb + SZ_X;
  float* qkvb  = tmpb + SZ_X;
  float* fg    = qkvb + SZ_Q;
  float* fu    = fg + SZ_F;

  const int nTok = MTOK;

  embed_kernel<<<nTok, D_, 0, stream>>>(tokens, embedding, x);

  for (int i = 0; i < 6; ++i) {
    // attention branch
    rmsnorm_kernel<<<nTok, D_, 0, stream>>>(x, L[i].norm1, xn);
    launch_gemm(0, xn, L[i].qkv, nullptr, nullptr, qkvb, nTok, 3 * D_, D_, stream);
    if (i == 5) {
      rope_kernel<<<(BB * NSEQ * H_ * 16 + 255) / 256, 256, 0, stream>>>(qkvb);
      flash_kernel<<<BB * H_ * (NSEQ / 16), 32, 0, stream>>>(qkvb, attn);
    } else {
      dsqg_kernel<<<(BB * H_ * NSEQ + 255) / 256, 256, 0, stream>>>(
          qkvb, L[i].pos_bias, attn);
    }
    launch_gemm(1, xn, L[i].gate_w, L[i].gate_b, nullptr, gateb, nTok, D_, D_,
                stream);
    mul_kernel<<<(int)((SZ_X + 255) / 256), 256, 0, stream>>>(attn, gateb,
                                                              (int)SZ_X);
    launch_gemm(3, attn, L[i].out, nullptr, x, x, nTok, D_, D_, stream);

    // interference block (layer 2 only)
    if (i == 2) {
      rmsnorm_kernel<<<nTok, D_, 0, stream>>>(x, L[i].inter_norm, xn);
      cumsum_pool_kernel<<<(BB * D_ + 255) / 256, 256, 0, stream>>>(xn, tmpb);
      launch_gemm(1, xn, L[i].ig_w, L[i].ig_b, nullptr, gateb, nTok, D_, D_,
                  stream);
      launch_gemm(4, tmpb, L[i].ip_w, L[i].ip_b, nullptr, attn, nTok, D_, D_,
                  stream);
      gated_add_kernel<<<(int)((SZ_X + 255) / 256), 256, 0, stream>>>(
          x, gateb, attn, (int)SZ_X);
    }

    // FFN: x += ffn_d( silu(h@ffn_g) * (h@ffn_u) )
    rmsnorm_kernel<<<nTok, D_, 0, stream>>>(x, L[i].norm2, xn);
    launch_gemm(2, xn, L[i].ffn_g, nullptr, nullptr, fg, nTok, FH_, D_, stream);
    launch_gemm(0, xn, L[i].ffn_u, nullptr, nullptr, fu, nTok, FH_, D_, stream);
    mul_kernel<<<(int)((SZ_F + 255) / 256), 256, 0, stream>>>(fg, fu, (int)SZ_F);
    launch_gemm(3, fg, L[i].ffn_d, nullptr, x, x, nTok, D_, FH_, stream);
  }

  // final norm + LM head (dominant GEMM: 4096 x 32000 x 256)
  rmsnorm_kernel<<<nTok, D_, 0, stream>>>(x, final_norm, xn);
  tdm_warm_kernel<<<1, 32, 0, stream>>>(lm_head);
  launch_gemm(0, xn, lm_head, nullptr, nullptr, (float*)d_out, nTok, VOCAB, D_,
              stream);
}
